// MobilityLoss_47296179863883
// MI455X (gfx1250) — compile-verified, matrix operand order checked by reading, not run on hardware
//
#include <hip/hip_runtime.h>
#include <hip/hip_bf16.h>

typedef __attribute__((ext_vector_type(16))) _Float16 v16h;
typedef __attribute__((ext_vector_type(8)))  float    v8f;

#define N_DIM 8192
#define K_DIM 256
#define NTHR  256                  // 8 waves per workgroup
#define WAVES (NTHR / 32)
#define RT    4                    // 16-row tiles per workgroup -> 64 rows
#define RBLK  (RT * 16)            // 64
#define NTILE (N_DIM / 16)         // 512 column tiles
#define NRB   (N_DIM / RBLK)       // 128 row blocks per direction
#define EPS   1e-4f

// One 32-K f16 fragment from a row-major f32 row.
// p = rowBase + kt*32 + hi*8 ; elems [0..7] <- p[0..7], [8..15] <- p[16..23]
__device__ __forceinline__ v16h load_frag_f32(const float* __restrict__ p) {
    v16h f;
#pragma unroll
    for (int i = 0; i < 8; ++i) {
        f[i]     = (_Float16)p[i];
        f[i + 8] = (_Float16)p[16 + i];
    }
    return f;
}

__global__ void zero_out_kernel(float* p) { p[0] = 0.0f; }

// f32 row-major (8192 x 256) -> f16 WMMA fragments [tile(512)][kt(8)][lane(32)][16]
__global__ __launch_bounds__(256)
void convert_frag_kernel(const float* __restrict__ src, _Float16* __restrict__ dst) {
    const int id   = blockIdx.x * 256 + threadIdx.x;   // (tile, kt, lane)
    const int lane = id & 31, kt = (id >> 5) & 7, tile = id >> 8;
    const int lr   = lane & 15, hi = lane >> 4;
    const float* p = src + (size_t)(tile * 16 + lr) * K_DIM + kt * 32 + hi * 8;
    *(v16h*)(dst + (size_t)id * 16) = load_frag_f32(p);
}

template <bool F16>
__global__ __launch_bounds__(NTHR)
void mobility_loss_kernel(const float* __restrict__ out1,
                          const float* __restrict__ out2,
                          const float* __restrict__ label,
                          const _Float16* __restrict__ h16_1,
                          const _Float16* __restrict__ h16_2,
                          float* __restrict__ loss_out)
{
    const int dir = blockIdx.x >> 7;        // 0: scores=out2@out1^T,label ; 1: swapped,label^T
    const int rb  = blockIdx.x & (NRB - 1); // 64-row block

    const float*    Qf = dir ? out1 : out2;
    const float*    Pf = dir ? out2 : out1;
    const _Float16* Qh = dir ? h16_1 : h16_2;
    const _Float16* Ph = dir ? h16_2 : h16_1;

    const int lane = threadIdx.x & 31;
    const int wave = threadIdx.x >> 5;
    const int hi   = lane >> 4;
    const int lr   = lane & 15;

    __shared__ _Float16 A_sh[RT * 8 * 32 * 16];   // 32 KB of A fragments (64 Q rows)
    __shared__ float red_m[WAVES][RBLK];
    __shared__ float red_s[WAVES][RBLK];
    __shared__ float lse_sh[RBLK];
    __shared__ float loss_red[WAVES];

    // ---- stage the workgroup's 64 Q rows (f16 fragment layout) into LDS ----
    for (int i = threadIdx.x; i < RT * 8 * 32; i += NTHR) {  // i = (rt*8+kt)*32+lane
        const int lf = i & 31, ktf = (i >> 5) & 7, rtf = i >> 8;
        v16h f;
        if (F16) {
            f = *(const v16h*)(Qh + ((size_t)((rb * RT + rtf) * 8 + ktf) * 32 + lf) * 16);
        } else {
            const int lrf = lf & 15, hif = lf >> 4;
            f = load_frag_f32(Qf + (size_t)(rb * RBLK + rtf * 16 + lrf) * K_DIM
                                 + ktf * 32 + hif * 8);
        }
        *(v16h*)(A_sh + (size_t)i * 16) = f;
    }
    __syncthreads();

    // One 16-col tile vs all 4 row-tiles. The empty asm re-defines `alo` every
    // call, so LICM cannot hoist the (loop-invariant) LDS A-fragment loads out
    // of the tile loop -- that hoist was spilling 256 VGPRs to scratch.
    auto compute_tile = [&](int t, v8f* c) {
        unsigned alo = 0;
        asm volatile("" : "+v"(alo));                 // opaque zero: blocks LICM/CSE
        const _Float16* abase = A_sh + alo;
        const v8f z = {};
#pragma unroll
        for (int rt = 0; rt < RT; ++rt) c[rt] = z;
#pragma unroll
        for (int kt = 0; kt < 8; ++kt) {
            v16h b;
            if (F16) {
                b = *(const v16h*)(Ph + ((size_t)(t * 8 + kt) * 32 + lane) * 16);
            } else {
                b = load_frag_f32(Pf + (size_t)(t * 16 + lr) * K_DIM + kt * 32 + hi * 8);
            }
#pragma unroll
            for (int rt = 0; rt < RT; ++rt) {
                const v16h a = *(const v16h*)(abase + (size_t)((rt * 8 + kt) * 32 + lane) * 16);
                c[rt] = __builtin_amdgcn_wmma_f32_16x16x32_f16(false, a, false, b,
                                                               (short)0, c[rt], false, false);
            }
        }
    };

    // ---- pass 1: per-lane online (max, sum-exp); no cross-lane work in hot loop ----
    float m[RT][8], s[RT][8];
#pragma unroll
    for (int rt = 0; rt < RT; ++rt)
#pragma unroll
        for (int v = 0; v < 8; ++v) { m[rt][v] = -__builtin_inff(); s[rt][v] = 0.0f; }

    for (int t = wave; t < NTILE; t += WAVES) {              // wave-uniform
        v8f c[RT];
        compute_tile(t, c);
#pragma unroll
        for (int rt = 0; rt < RT; ++rt)
#pragma unroll
            for (int v = 0; v < 8; ++v) {
                const float x  = c[rt][v];                   // row rt*16+v+8*hi, col t*16+lr
                const float nm = fmaxf(m[rt][v], x);
                s[rt][v] = s[rt][v] * __expf(m[rt][v] - nm) + __expf(x - nm);
                m[rt][v] = nm;
            }
    }

    // combine (m,s) partials across the 16-lane half, then across waves via LDS
#pragma unroll
    for (int rt = 0; rt < RT; ++rt)
#pragma unroll
        for (int v = 0; v < 8; ++v) {
            float mm = m[rt][v], ss = s[rt][v];
#pragma unroll
            for (int mask = 1; mask <= 8; mask <<= 1) {
                const float mo = __shfl_xor(mm, mask);
                const float so = __shfl_xor(ss, mask);
                const float M  = fmaxf(mm, mo);
                ss = ss * __expf(mm - M) + so * __expf(mo - M);
                mm = M;
            }
            if (lr == 0) {
                red_m[wave][rt * 16 + v + 8 * hi] = mm;
                red_s[wave][rt * 16 + v + 8 * hi] = ss;
            }
        }
    __syncthreads();
    if (threadIdx.x < RBLK) {
        const int r = threadIdx.x;
        float M = -__builtin_inff();
        for (int w = 0; w < WAVES; ++w) M = fmaxf(M, red_m[w][r]);
        float S = 0.0f;
        for (int w = 0; w < WAVES; ++w) S += red_s[w][r] * __expf(red_m[w][r] - M);
        lse_sh[r] = M + __logf(S);                           // per-row log-sum-exp
    }
    __syncthreads();

    float lse[RT][8];
#pragma unroll
    for (int rt = 0; rt < RT; ++rt)
#pragma unroll
        for (int v = 0; v < 8; ++v) lse[rt][v] = lse_sh[rt * 16 + v + 8 * hi];

    // ---- pass 2: recompute tiles, accumulate -label * log(softmax + eps) ----
    float acc = 0.0f;
    for (int t = wave; t < NTILE; t += WAVES) {
        v8f c[RT];
        compute_tile(t, c);
        const int col = t * 16 + lr;
#pragma unroll
        for (int rt = 0; rt < RT; ++rt)
#pragma unroll
            for (int v = 0; v < 8; ++v) {
                const int row = rb * RBLK + rt * 16 + v + 8 * hi;
                const size_t idx = (dir == 0) ? (size_t)row * N_DIM + col
                                              : (size_t)col * N_DIM + row;
                const float ph = __expf(c[rt][v] - lse[rt][v]);
                acc -= label[idx] * __logf(ph + EPS);
            }
    }

    // ---- block reduction, one f32 atomic per workgroup ----
    acc += __shfl_xor(acc, 1);
    acc += __shfl_xor(acc, 2);
    acc += __shfl_xor(acc, 4);
    acc += __shfl_xor(acc, 8);
    acc += __shfl_xor(acc, 16);
    if (lane == 0) loss_red[wave] = acc;
    __syncthreads();
    if (threadIdx.x == 0) {
        float tot = 0.0f;
#pragma unroll
        for (int w = 0; w < WAVES; ++w) tot += loss_red[w];
        atomicAdd(loss_out, tot);
    }
}

extern "C" void kernel_launch(void* const* d_in, const int* in_sizes, int n_in,
                              void* d_out, int out_size, void* d_ws, size_t ws_size,
                              hipStream_t stream) {
    (void)in_sizes; (void)n_in; (void)out_size;
    const float* out1  = (const float*)d_in[0];
    const float* out2  = (const float*)d_in[1];
    const float* label = (const float*)d_in[2];
    float* out = (float*)d_out;

    zero_out_kernel<<<1, 1, 0, stream>>>(out);

    const size_t halfElems = (size_t)N_DIM * K_DIM;              // per matrix
    const size_t need      = 2 * halfElems * sizeof(_Float16);   // 8 MB

    if (ws_size >= need) {
        _Float16* h1 = (_Float16*)d_ws;
        _Float16* h2 = h1 + halfElems;
        const int convBlocks = (NTILE * 8 * 32) / 256;           // 512
        convert_frag_kernel<<<convBlocks, 256, 0, stream>>>(out1, h1);
        convert_frag_kernel<<<convBlocks, 256, 0, stream>>>(out2, h2);
        mobility_loss_kernel<true><<<2 * NRB, NTHR, 0, stream>>>(out1, out2, label,
                                                                 h1, h2, out);
    } else {
        mobility_loss_kernel<false><<<2 * NRB, NTHR, 0, stream>>>(out1, out2, label,
                                                                  nullptr, nullptr, out);
    }
}
